// Model_58463094833839
// MI455X (gfx1250) — compile-verified, matrix operand order checked by reading, not run on hardware
//
#include <hip/hip_runtime.h>

#define NN 100000
#define NE1 600000
#define NE2 300000
#define HD 128
#define INF 35
#define NG 1000

typedef __attribute__((ext_vector_type(16))) __bf16 v16bf;
typedef __attribute__((ext_vector_type(8)))  float  v8f;
typedef __attribute__((ext_vector_type(4)))  unsigned int v4u;
typedef __attribute__((ext_vector_type(8)))  int v8i_;
typedef __attribute__((ext_vector_type(4)))  int v4i_;

struct U16x16 { unsigned short v[16]; };

__device__ __forceinline__ unsigned short f2bf(float f) {
    unsigned int u = __float_as_uint(f);
    unsigned int r = u + 0x7FFFu + ((u >> 16) & 1u);
    return (unsigned short)(r >> 16);
}

__device__ __forceinline__ void atomAddF(float* p, float v) {
    __hip_atomic_fetch_add(p, v, __ATOMIC_RELAXED, __HIP_MEMORY_SCOPE_AGENT);
}

__device__ __forceinline__ float siluf(float x) {
    return x * (1.0f / (1.0f + __expf(-x)));
}

// ---- pack nodeW [6][128][128] f32 row-major -> bf16 WMMA B-fragments ----
// fragment f = kt*8+nt covers K in [kt*32,kt*32+32), Ncol in [nt*16,nt*16+16)
// lane<16 holds col n=lane, k-locals {0..7,16..23}; lane>=16 holds {8..15,24..31}
__global__ void k_pack_w(const float* __restrict__ nodeW, unsigned short* __restrict__ wpk) {
    int tid = blockIdx.x * blockDim.x + threadIdx.x;
    if (tid >= 6 * 32 * 32 * 16) return;
    int i    = tid & 15;
    int lane = (tid >> 4) & 31;
    int f    = (tid >> 9) & 31;
    int mat  = tid >> 14;
    int kt = f >> 3, nt = f & 7;
    int n  = nt * 16 + (lane & 15);
    int kl = (lane < 16) ? ((i < 8) ? i : 8 + i) : ((i < 8) ? 8 + i : 16 + i);
    int k  = kt * 32 + kl;
    wpk[tid] = f2bf(nodeW[((size_t)mat * HD + k) * HD + n]);
}

// ---- h = silu(x @ W_in + b_in) : one block (128 thr) per node ----
__global__ void k_embed(const float* __restrict__ x, const float* __restrict__ Wi,
                        const float* __restrict__ bi, float* __restrict__ h) {
    int n = blockIdx.x, c = threadIdx.x;
    __shared__ float xr[INF];
    if (c < INF) xr[c] = x[(size_t)n * INF + c];
    __syncthreads();
    float acc = bi[c];
#pragma unroll
    for (int k = 0; k < INF; ++k) acc = fmaf(xr[k], Wi[k * HD + c], acc);
    h[(size_t)n * HD + c] = siluf(acc);
}

// ---- per-edge: radial = silu(rbf@cW+cB), msg = h[row]*radial, scatter-add ----
__global__ void k_edge(const float* __restrict__ h, const float* __restrict__ pos,
                       const int* __restrict__ ei, int E,
                       const float* __restrict__ cW, const float* __restrict__ cB,
                       float* __restrict__ agg) {
    int gid = blockIdx.x * blockDim.x + threadIdx.x;
    int e = gid >> 5, lane = gid & 31;
    if (e >= E) return;
    int row = ei[e], col = ei[E + e];
    float dx = pos[row * 3 + 0] - pos[col * 3 + 0];
    float dy = pos[row * 3 + 1] - pos[col * 3 + 1];
    float dz = pos[row * 3 + 2] - pos[col * 3 + 2];
    float d = sqrtf(fmaf(dx, dx, fmaf(dy, dy, dz * dz)));
    float rbf[9];
#pragma unroll
    for (int i = 0; i < 9; ++i) {
        float t = (d - 0.75f * (float)i) * 1.5f;   // sigma = 2/3
        rbf[i] = __expf(-t * t);
    }
    int c0 = lane * 4;
    float4 acc = *(const float4*)(cB + c0);
#pragma unroll
    for (int i = 0; i < 9; ++i) {
        float4 w = *(const float4*)(cW + i * HD + c0);
        acc.x = fmaf(rbf[i], w.x, acc.x);
        acc.y = fmaf(rbf[i], w.y, acc.y);
        acc.z = fmaf(rbf[i], w.z, acc.z);
        acc.w = fmaf(rbf[i], w.w, acc.w);
    }
    float4 hv = *(const float4*)(h + (size_t)row * HD + c0);
    float* ap = agg + (size_t)col * HD + c0;
    atomAddF(ap + 0, hv.x * siluf(acc.x));
    atomAddF(ap + 1, hv.y * siluf(acc.y));
    atomAddF(ap + 2, hv.z * siluf(acc.z));
    atomAddF(ap + 3, hv.w * siluf(acc.w));
}

// ---- build one bf16 A-fragment in registers from z = h+agg (fused convert) ----
__device__ __forceinline__ v16bf loadA(const float* __restrict__ h,
                                       const float* __restrict__ agg,
                                       int row, int e0) {
    const float* hp = h + (size_t)row * HD;
    const float* gp = agg + (size_t)row * HD;
    float4 h0 = *(const float4*)(hp + e0);
    float4 h1 = *(const float4*)(hp + e0 + 4);
    float4 h2 = *(const float4*)(hp + e0 + 16);
    float4 h3 = *(const float4*)(hp + e0 + 20);
    float4 g0 = *(const float4*)(gp + e0);
    float4 g1 = *(const float4*)(gp + e0 + 4);
    float4 g2 = *(const float4*)(gp + e0 + 16);
    float4 g3 = *(const float4*)(gp + e0 + 20);
    U16x16 t;
    t.v[0]  = f2bf(h0.x + g0.x); t.v[1]  = f2bf(h0.y + g0.y);
    t.v[2]  = f2bf(h0.z + g0.z); t.v[3]  = f2bf(h0.w + g0.w);
    t.v[4]  = f2bf(h1.x + g1.x); t.v[5]  = f2bf(h1.y + g1.y);
    t.v[6]  = f2bf(h1.z + g1.z); t.v[7]  = f2bf(h1.w + g1.w);
    t.v[8]  = f2bf(h2.x + g2.x); t.v[9]  = f2bf(h2.y + g2.y);
    t.v[10] = f2bf(h2.z + g2.z); t.v[11] = f2bf(h2.w + g2.w);
    t.v[12] = f2bf(h3.x + g3.x); t.v[13] = f2bf(h3.y + g3.y);
    t.v[14] = f2bf(h3.z + g3.z); t.v[15] = f2bf(h3.w + g3.w);
    return __builtin_bit_cast(v16bf, t);
}

// ---- node update: out = bn(leaky((h+agg) @ nW + nB)) ----
// 5 waves/block, each wave owns 32 rows (two 16-row M-tiles) x 128 cols.
// Weights (32KB packed bf16 fragments) staged into LDS via the Tensor Data
// Mover; B fragments then read with ds_load_b128 and reused by 2 WMMAs.
__global__ void __launch_bounds__(160) k_node(
        const float* __restrict__ h, const float* __restrict__ agg,
        const unsigned short* __restrict__ wpk,
        const float* __restrict__ nb, const float* __restrict__ g,
        const float* __restrict__ bb, float* __restrict__ hout, int accum) {
    __shared__ unsigned short smem[16384];   // 32KB packed weight fragments

#if __has_builtin(__builtin_amdgcn_tensor_load_to_lds)
    if (threadIdx.x < 32) {   // wave 0 of the block issues one TDM descriptor
        unsigned long long wa = (unsigned long long)(uintptr_t)wpk;
        unsigned lb = (unsigned)(uintptr_t)(void*)smem;   // LDS byte offset
        v4u g0 = { 1u,                                    // count=1
                   lb,                                    // lds_addr
                   (unsigned)(wa & 0xFFFFFFFFu),          // global_addr[31:0]
                   (unsigned)((wa >> 32) & 0x01FFFFFFu) | 0x80000000u }; // addr[56:32] | type=2
        v8i_ g1 = { (int)(3u << 16),        // data_size = 8B elements
                    (int)(4096u << 16),     // tensor_dim0 = 4096 (x8B = 32KB)
                    (int)(1u << 16),        // tensor_dim1 = 1
                    (int)(4096u << 16),     // tile_dim0 = 4096
                    1,                      // tile_dim1 = 1, tile_dim2 = 0
                    4096,                   // tensor_dim0_stride
                    0, 0 };
        v4i_ gz4 = { 0, 0, 0, 0 };
        v8i_ gz8 = { 0, 0, 0, 0, 0, 0, 0, 0 };
        __builtin_amdgcn_tensor_load_to_lds(g0, g1, gz4, gz4, gz8, 0);
        __builtin_amdgcn_s_wait_tensorcnt(0);
    }
#else
    for (int i = threadIdx.x; i < 8192; i += blockDim.x)
        ((unsigned int*)smem)[i] = ((const unsigned int*)wpk)[i];
#endif
    __syncthreads();

    int lane = threadIdx.x & 31;
    int wave = blockIdx.x * 5 + (threadIdx.x >> 5);
    int rowbase = wave * 32;              // 3125 waves * 32 rows = 100000 exactly

    v8f acc0[8] = {};
    v8f acc1[8] = {};
    int mrow = rowbase + (lane & 15);
    const char* sb = (const char*)smem;

#pragma unroll
    for (int kt = 0; kt < 4; ++kt) {
        int e0 = kt * 32 + ((lane & 16) ? 8 : 0);
        v16bf a0 = loadA(h, agg, mrow,      e0);
        v16bf a1 = loadA(h, agg, mrow + 16, e0);
#pragma unroll
        for (int nt = 0; nt < 8; ++nt) {
            int f = kt * 8 + nt;
            struct BF { uint4 lo, hi; } bfrag;
            bfrag.lo = *(const uint4*)(sb + f * 1024 + lane * 32);
            bfrag.hi = *(const uint4*)(sb + f * 1024 + lane * 32 + 16);
            v16bf bm = __builtin_bit_cast(v16bf, bfrag);
            acc0[nt] = __builtin_amdgcn_wmma_f32_16x16x32_bf16(
                false, a0, false, bm, (short)0, acc0[nt], false, false);
            acc1[nt] = __builtin_amdgcn_wmma_f32_16x16x32_bf16(
                false, a1, false, bm, (short)0, acc1[nt], false, false);
        }
    }

    const float inv = 0.99999500003750f;   // 1/sqrt(1+1e-5)
    int cbase = lane & 15;
    int roff  = (lane & 16) ? 8 : 0;
#pragma unroll
    for (int nt = 0; nt < 8; ++nt) {
        int col = nt * 16 + cbase;
        float bias = nb[col], gg = g[col], bv = bb[col];
#pragma unroll
        for (int r = 0; r < 8; ++r) {
            float v0 = acc0[nt][r] + bias;
            v0 = (v0 >= 0.0f) ? v0 : 0.01f * v0;
            v0 = gg * v0 * inv + bv;
            float v1 = acc1[nt][r] + bias;
            v1 = (v1 >= 0.0f) ? v1 : 0.01f * v1;
            v1 = gg * v1 * inv + bv;
            float* d0 = hout + (size_t)(rowbase + roff + r) * HD + col;
            float* d1 = hout + (size_t)(rowbase + 16 + roff + r) * HD + col;
            if (accum) { *d0 += v0; *d1 += v1; }
            else       { *d0  = v0; *d1  = v1; }
        }
    }
}

// ---- global_add_pool: atomics into hg[G,128] ----
__global__ void k_pool(const float* __restrict__ h, const int* __restrict__ batch,
                       float* __restrict__ hg) {
    int id = blockIdx.x * blockDim.x + threadIdx.x;
    if (id >= NN * 32) return;
    int n = id >> 5, c4 = (id & 31) * 4;
    int gi = batch[n];
    float4 v = *(const float4*)(h + (size_t)n * HD + c4);
    float* dst = hg + (size_t)gi * HD + c4;
    atomAddF(dst + 0, v.x);
    atomAddF(dst + 1, v.y);
    atomAddF(dst + 2, v.z);
    atomAddF(dst + 3, v.w);
}

// ---- fc: out = bn(leaky(in @ W + b)); one block (128 thr) per graph ----
__global__ void k_fc(const float* __restrict__ in, const float* __restrict__ W,
                     const float* __restrict__ bias, const float* __restrict__ g,
                     const float* __restrict__ bb, float* __restrict__ out) {
    int gi = blockIdx.x, c = threadIdx.x;
    __shared__ float row[HD];
    row[c] = in[(size_t)gi * HD + c];
    __syncthreads();
    float acc = bias[c];
#pragma unroll 8
    for (int k = 0; k < HD; ++k) acc = fmaf(row[k], W[k * HD + c], acc);
    acc = (acc >= 0.0f) ? acc : 0.01f * acc;
    out[(size_t)gi * HD + c] = g[c] * acc * 0.99999500003750f + bb[c];
}

// ---- head: out[g] = hg[g,:] @ outW + outB ----
__global__ void k_out(const float* __restrict__ hg, const float* __restrict__ ow,
                      const float* __restrict__ ob, float* __restrict__ out) {
    int gi = blockIdx.x * blockDim.x + threadIdx.x;
    if (gi >= NG) return;
    float acc = ob[0];
#pragma unroll 8
    for (int k = 0; k < HD; ++k) acc = fmaf(hg[(size_t)gi * HD + k], ow[k], acc);
    out[gi] = acc;
}

extern "C" void kernel_launch(void* const* d_in, const int* in_sizes, int n_in,
                              void* d_out, int out_size, void* d_ws, size_t ws_size,
                              hipStream_t stream) {
    (void)in_sizes; (void)n_in; (void)out_size; (void)ws_size;
    const float* x      = (const float*)d_in[0];
    const float* pos    = (const float*)d_in[1];
    const float* W_in   = (const float*)d_in[2];
    const float* b_in   = (const float*)d_in[3];
    const float* coordW = (const float*)d_in[4];   // [3,2,9,128]
    const float* coordB = (const float*)d_in[5];   // [3,2,128]
    const float* nodeW  = (const float*)d_in[6];   // [3,2,128,128]
    const float* nodeB  = (const float*)d_in[7];   // [3,2,128]
    const float* bnG    = (const float*)d_in[8];
    const float* bnB    = (const float*)d_in[9];
    const float* fcW    = (const float*)d_in[10];  // [3,128,128]
    const float* fcB    = (const float*)d_in[11];
    const float* fcBnG  = (const float*)d_in[12];
    const float* fcBnB  = (const float*)d_in[13];
    const float* outW   = (const float*)d_in[14];
    const float* outB   = (const float*)d_in[15];
    const int*   ei1    = (const int*)d_in[16];
    const int*   ei2    = (const int*)d_in[17];
    const int*   batch  = (const int*)d_in[18];
    float* out = (float*)d_out;

    char* ws = (char*)d_ws;
    size_t off = 0;
    auto alloc = [&](size_t bytes) -> void* {
        void* p = ws + off;
        off += bytes;
        off = (off + 255) & ~(size_t)255;
        return p;
    };
    float*          hA  = (float*)alloc((size_t)NN * HD * 4);
    float*          hB  = (float*)alloc((size_t)NN * HD * 4);
    float*          agg = (float*)alloc((size_t)NN * HD * 4);
    unsigned short* wpk = (unsigned short*)alloc((size_t)6 * 32 * 32 * 16 * 2);
    float*          hg0 = (float*)alloc((size_t)NG * HD * 4);
    float*          hg1 = (float*)alloc((size_t)NG * HD * 4);

    k_pack_w<<<(6 * 32 * 32 * 16 + 255) / 256, 256, 0, stream>>>(nodeW, wpk);
    k_embed<<<NN, HD, 0, stream>>>(x, W_in, b_in, hA);

    float* hcur = hA;
    float* hnext = hB;

    for (int l = 0; l < 3; ++l) {
        for (int br = 0; br < 2; ++br) {
            const int* ei = br ? ei2 : ei1;
            int E = br ? NE2 : NE1;
            int mat = l * 2 + br;
            (void)hipMemsetAsync(agg, 0, (size_t)NN * HD * 4, stream);
            k_edge<<<(E * 32 + 255) / 256, 256, 0, stream>>>(
                hcur, pos, ei, E,
                coordW + (size_t)mat * 9 * HD, coordB + (size_t)mat * HD, agg);
            // 625 blocks x 5 waves x 32 rows = 100000 rows exactly (no tail)
            k_node<<<625, 160, 0, stream>>>(
                hcur, agg, wpk + (size_t)mat * (32 * 32 * 16),
                nodeB + (size_t)mat * HD, bnG + (size_t)mat * HD, bnB + (size_t)mat * HD,
                hnext, br);
        }
        float* t = hcur; hcur = hnext; hnext = t;
    }

    (void)hipMemsetAsync(hg0, 0, (size_t)NG * HD * 4, stream);
    k_pool<<<(NN * 32 + 255) / 256, 256, 0, stream>>>(hcur, batch, hg0);

    k_fc<<<NG, HD, 0, stream>>>(hg0, fcW + 0 * HD * HD, fcB + 0 * HD,
                                fcBnG + 0 * HD, fcBnB + 0 * HD, hg1);
    k_fc<<<NG, HD, 0, stream>>>(hg1, fcW + 1 * HD * HD, fcB + 1 * HD,
                                fcBnG + 1 * HD, fcBnB + 1 * HD, hg0);
    k_fc<<<NG, HD, 0, stream>>>(hg0, fcW + 2 * HD * HD, fcB + 2 * HD,
                                fcBnG + 2 * HD, fcBnB + 2 * HD, hg1);

    k_out<<<(NG + 127) / 128, 128, 0, stream>>>(hg1, outW, outB, out);
}
